// MoEBigConv_12644383719774
// MI455X (gfx1250) — matrix operand order repaired and checked by reading, not compiled, not run on hardware
//
#include <hip/hip_runtime.h>
#include <hip/hip_bf16.h>

// ---------------------------------------------------------------------------
// MI455X (gfx1250) implementation of MoEBigConv.
//   * All GEMM-shaped ops (MoE conv via im2col, routers, expert linears) run
//     on v_wmma_f32_16x16x32_f16 (wave32, fp32 accum). A block of 4 waves
//     computes a 64x16 C macro-tile; the shared 16x128 B (weight) chunk is
//     streamed into LDS by the Tensor Data Mover (tensor_load_to_lds,
//     TENSORcnt double-buffering), A streams from global with prefetch.
//   * Front convs (K=27/512/288, ~20% of FLOPs) are direct fp32 VALU.
//   * Top-k channel selection is an exact rank-count reimplementation of
//     jax.lax.top_k + sorted gather.
// ---------------------------------------------------------------------------

typedef __attribute__((ext_vector_type(16))) _Float16 v16h;
typedef __attribute__((ext_vector_type(8)))  _Float16 v8h;
typedef __attribute__((ext_vector_type(8)))  float    v8f;
typedef __attribute__((ext_vector_type(4)))  unsigned int u32x4;
typedef __attribute__((ext_vector_type(8)))  int      i32x8;
typedef __attribute__((ext_vector_type(4)))  int      i32x4;

#define GEMM_PLAIN 0
#define GEMM_ABS   1

#if __has_builtin(__builtin_amdgcn_tensor_load_to_lds)
#define HAVE_TDM 1
#else
#define HAVE_TDM 0
#endif

#if HAVE_TDM
// Issue a TDM load of a 2-D f16 tile: 16 rows x 128 cols, row stride = K
// elements, from gptr (tile start) into LDS byte-offset lds_off.
// D# packing per CDNA5 ISA ch.8 (group0 128b, group1 256b; groups 2/3 unused).
__device__ __forceinline__ void tdm_load_b16_tile(const _Float16* gptr,
                                                  unsigned lds_off,
                                                  int K, int Nrows) {
  unsigned long long ga = (unsigned long long)(uintptr_t)gptr;
  u32x4 g0;
  g0[0] = 1u;                                            // count=1 (valid), user D#
  g0[1] = lds_off;                                       // lds_addr (bytes)
  g0[2] = (unsigned)(ga & 0xffffffffu);                  // global_addr[31:0]
  g0[3] = (unsigned)((ga >> 32) & 0x01ffffffu) | (2u << 30); // addr[56:32] | type=2
  i32x8 g1;
  unsigned K_u = (unsigned)K, N_u = (unsigned)Nrows;
  g1[0] = (int)(1u << 16);                               // data_size=1 (2 bytes)
  g1[1] = (int)((K_u & 0xffffu) << 16);                  // tensor_dim0[15:0] @ bits63:48
  g1[2] = (int)((K_u >> 16) | ((N_u & 0xffffu) << 16));  // dim0[31:16] | dim1[15:0]
  g1[3] = (int)((N_u >> 16) | (128u << 16));             // dim1[31:16] | tile_dim0=128
  g1[4] = (int)16u;                                      // tile_dim1=16, tile_dim2=0
  g1[5] = (int)K_u;                                      // tensor_dim0_stride[31:0]
  g1[6] = 0;                                             // stride[47:32] | dim1_stride lo
  g1[7] = 0;
  i32x4 g2 = {0, 0, 0, 0};
  i32x4 g3 = {0, 0, 0, 0};
#if __clang_major__ >= 23
  i32x8 g4 = {0, 0, 0, 0, 0, 0, 0, 0};
  __builtin_amdgcn_tensor_load_to_lds(g0, g1, g2, g3, g4, 0);
#else
  __builtin_amdgcn_tensor_load_to_lds(g0, g1, g2, g3, 0);
#endif
}
#endif  // HAVE_TDM

// ---------------------------------------------------------------------------
// conv0: normalize + 3x3 s1 p1, 3->32ch, 32x32, ReLU. Zero-padding applies to
// the *normalized* input (matches reference), so normalization is per-tap.
// ---------------------------------------------------------------------------
__global__ void conv0_kernel(const float* __restrict__ x,
                             const float* __restrict__ W0,
                             const float* __restrict__ b0,
                             float* __restrict__ out, int total) {
  int idx = blockIdx.x * blockDim.x + threadIdx.x;
  if (idx >= total) return;
  const float mean[3] = {0.4914f, 0.4822f, 0.4465f};
  const float inv[3]  = {1.0f/0.2023f, 1.0f/0.1994f, 1.0f/0.201f};
  int ox = idx & 31, oy = (idx >> 5) & 31, oc = (idx >> 10) & 31, b = idx >> 15;
  const float* wp = W0 + oc * 27;
  float acc = b0[oc];
  for (int ci = 0; ci < 3; ++ci) {
    const float* xp = x + ((size_t)(b * 3 + ci)) * 1024;
    float mu = mean[ci], iv = inv[ci];
    #pragma unroll
    for (int ky = 0; ky < 3; ++ky) {
      int iy = oy + ky - 1;
      if ((unsigned)iy < 32u) {
        #pragma unroll
        for (int kx = 0; kx < 3; ++kx) {
          int ix = ox + kx - 1;
          if ((unsigned)ix < 32u)
            acc += wp[ci * 9 + ky * 3 + kx] * ((xp[iy * 32 + ix] - mu) * iv);
        }
      }
    }
  }
  out[idx] = fmaxf(acc, 0.f);
}

// conv2: 4x4 s2 p1, 32->32ch, 32x32 -> 16x16, ReLU
__global__ void conv2_kernel(const float* __restrict__ in,
                             const float* __restrict__ W,
                             const float* __restrict__ bias,
                             float* __restrict__ out, int total) {
  int idx = blockIdx.x * blockDim.x + threadIdx.x;
  if (idx >= total) return;
  int ox = idx & 15, oy = (idx >> 4) & 15, oc = (idx >> 8) & 31, b = idx >> 13;
  float acc = bias[oc];
  for (int ci = 0; ci < 32; ++ci) {
    const float* xp = in + ((size_t)(b * 32 + ci)) * 1024;
    const float* wp = W + (oc * 32 + ci) * 16;
    #pragma unroll
    for (int ky = 0; ky < 4; ++ky) {
      int iy = oy * 2 - 1 + ky;
      if ((unsigned)iy < 32u) {
        #pragma unroll
        for (int kx = 0; kx < 4; ++kx) {
          int ix = ox * 2 - 1 + kx;
          if ((unsigned)ix < 32u) acc += wp[ky * 4 + kx] * xp[iy * 32 + ix];
        }
      }
    }
  }
  out[idx] = fmaxf(acc, 0.f);
}

// conv4: 3x3 s1 p1, 32->64ch, 16x16, ReLU
__global__ void conv4_kernel(const float* __restrict__ in,
                             const float* __restrict__ W,
                             const float* __restrict__ bias,
                             float* __restrict__ out, int total) {
  int idx = blockIdx.x * blockDim.x + threadIdx.x;
  if (idx >= total) return;
  int ox = idx & 15, oy = (idx >> 4) & 15, oc = (idx >> 8) & 63, b = idx >> 14;
  float acc = bias[oc];
  for (int ci = 0; ci < 32; ++ci) {
    const float* xp = in + ((size_t)(b * 32 + ci)) * 256;
    const float* wp = W + (oc * 32 + ci) * 9;
    #pragma unroll
    for (int ky = 0; ky < 3; ++ky) {
      int iy = oy - 1 + ky;
      if ((unsigned)iy < 16u) {
        #pragma unroll
        for (int kx = 0; kx < 3; ++kx) {
          int ix = ox - 1 + kx;
          if ((unsigned)ix < 16u) acc += wp[ky * 3 + kx] * xp[iy * 16 + ix];
        }
      }
    }
  }
  out[idx] = fmaxf(acc, 0.f);
}

// ---------------------------------------------------------------------------
// Conversions
// ---------------------------------------------------------------------------
__global__ void f32_to_f16_kernel(const float* __restrict__ src,
                                  _Float16* __restrict__ dst, int n) {
  int i = blockIdx.x * blockDim.x + threadIdx.x;
  if (i < n) dst[i] = (_Float16)src[i];
}

// dst[n][k] = (n < N) ? src[k][n] : 0   (transpose [K,N] f32 -> [Npad,K] f16)
__global__ void transpose_pad_f16_kernel(const float* __restrict__ src,
                                         _Float16* __restrict__ dst,
                                         int K, int N, int Npad) {
  int i = blockIdx.x * blockDim.x + threadIdx.x;
  int total = Npad * K;
  if (i >= total) return;
  int n = i / K, k = i - n * K;
  dst[i] = (n < N) ? (_Float16)src[(size_t)k * N + n] : (_Float16)0.f;
}

// row-pad: [Nrows,K] f32 -> [Npad,K] f16 (zeros in padded rows)
__global__ void pad_rows_f16_kernel(const float* __restrict__ src,
                                    _Float16* __restrict__ dst,
                                    int K, int Nrows, int Npad) {
  int i = blockIdx.x * blockDim.x + threadIdx.x;
  int total = Npad * K;
  if (i >= total) return;
  int n = i / K, k = i - n * K;
  dst[i] = (n < Nrows) ? (_Float16)src[(size_t)n * K + k] : (_Float16)0.f;
}

// im2col for MoE conv (4x4 s2 p1 on [B,64,16,16] -> col[m=b*64+oy*8+ox][k=ci*16+ky*4+kx])
__global__ void im2col6_kernel(const float* __restrict__ a3,
                               _Float16* __restrict__ col, int total) {
  int idx = blockIdx.x * blockDim.x + threadIdx.x;
  if (idx >= total) return;
  int k = idx & 1023, m = idx >> 10;
  int ox = m & 7, oy = (m >> 3) & 7, b = m >> 6;
  int kx = k & 3, ky = (k >> 2) & 3, ci = k >> 4;
  int iy = oy * 2 - 1 + ky, ix = ox * 2 - 1 + kx;
  float v = 0.f;
  if ((unsigned)iy < 16u && (unsigned)ix < 16u)
    v = a3[(((size_t)(b * 64 + ci)) * 16 + iy) * 16 + ix];
  col[idx] = (_Float16)v;
}

// ---------------------------------------------------------------------------
// WMMA GEMM:  C[M,Npad] f32 = A[M,K] f16 (row-major) x W[Npad,K] f16
// (row-major, i.e. B = W^T) + bias, optional |.| for routers.
// Block = 4 waves -> 64x16 C macro-tile (wave w handles rows mtb*64+16w..+15).
// The shared B chunk (16 rows x 128 k, 4KB) is TDM-DMA'd into LDS, ping-pong
// double buffered on TENSORcnt; A fragments load straight from global
// (CDNA5 ISA 7.12.2 layouts make both sides contiguous vector loads).
// Requires: M % 64 == 0, Npad % 16 == 0, K % 128 == 0 (true for all stages).
// ---------------------------------------------------------------------------
__global__ void __launch_bounds__(128)
wmma_gemm_kernel(const _Float16* __restrict__ A, const _Float16* __restrict__ W,
                 const float* __restrict__ bias, float* __restrict__ C,
                 int M, int Npad, int Nbias, int K, int mode) {
  __shared__ _Float16 sB[2][16 * 128];

  int nT = Npad >> 4;
  int mtb = blockIdx.x / nT;
  int nt = blockIdx.x - mtb * nT;
  int wave = threadIdx.x >> 5;
  int lane = threadIdx.x & 31;
  int r = lane & 15, g = lane >> 4;
  int m0 = mtb * 64 + wave * 16;

  const _Float16* __restrict__ arow  = A + (size_t)(m0 + r) * K;
  const _Float16* __restrict__ wbase = W + (size_t)(nt * 16) * K;

  int nChunks = K >> 7;
  v8f acc = {};

#if HAVE_TDM
  if (threadIdx.x < 32)  // wave 0 drives the DMA; EXEC ignored by tensor ops
    tdm_load_b16_tile(wbase, (unsigned)(uintptr_t)&sB[0][0], K, Npad);
#endif

  for (int c = 0; c < nChunks; ++c) {
#if HAVE_TDM
    if (threadIdx.x < 32) {
      if (c + 1 < nChunks) {
        tdm_load_b16_tile(wbase + (size_t)(c + 1) * 128,
                          (unsigned)(uintptr_t)&sB[(c + 1) & 1][0], K, Npad);
        __builtin_amdgcn_s_wait_tensorcnt(1);  // chunk c landed, c+1 in flight
      } else {
        __builtin_amdgcn_s_wait_tensorcnt(0);
      }
    }
#else
    // Fallback: cooperative global->LDS staging of chunk c.
    for (int t = threadIdx.x; t < 16 * 128; t += 128) {
      int rr = t >> 7, kk = t & 127;
      sB[c & 1][t] = wbase[(size_t)rr * K + (size_t)c * 128 + kk];
    }
#endif
    __syncthreads();  // B chunk c visible to all 4 waves

    const _Float16* sb = &sB[c & 1][r * 128 + g * 16];
    int ktb = c << 7;
    #pragma unroll
    for (int u = 0; u < 4; ++u) {
      int kt = ktb + u * 32;
      if (kt + 256 < K)
        __builtin_prefetch((const void*)(arow + kt + 256), 0, 1);
      v8h alo = *(const v8h*)(arow + kt + g * 16);
      v8h ahi = *(const v8h*)(arow + kt + g * 16 + 8);
      v16h af, bf;
      #pragma unroll
      for (int i = 0; i < 8; ++i) { af[i] = alo[i]; af[i + 8] = ahi[i]; }
      bf = *(const v16h*)(sb + u * 32);
      acc = __builtin_amdgcn_wmma_f32_16x16x32_f16(
          /*neg_a=*/false, af, /*neg_b=*/false, bf,
          /*c_mod=*/(short)0, acc, /*reuse_a=*/false, /*reuse_b=*/false);
    }
    __syncthreads();  // all reads of this buffer done before TDM reuses it
  }

  int n = nt * 16 + r;
  float bv = (bias && n < Nbias) ? bias[n] : 0.f;
  size_t base = (size_t)(m0 + g * 8) * Npad + n;
  #pragma unroll
  for (int v = 0; v < 8; ++v) {
    float val = acc[v] + bv;
    if (mode == GEMM_ABS) val = fabsf(val);
    C[base + (size_t)v * Npad] = val;
  }
}

// ---------------------------------------------------------------------------
// Top-k channel selection (exact top_k + sorted-index gather semantics).
// rank_i = #{j : s_j>s_i or (==, j<i)}; keep rank<keep; out pos = #kept j<i.
// ---------------------------------------------------------------------------
__global__ void select_linear_kernel(const float* __restrict__ full,
                                     const float* __restrict__ scores,
                                     int N, int Npad, int keep, int do_relu,
                                     _Float16* __restrict__ out_h,
                                     float* __restrict__ out_f) {
  int row = blockIdx.x;
  __shared__ float s[2048];
  __shared__ int   kp[2048];
  const float* sr = scores + (size_t)row * Npad;
  for (int i = threadIdx.x; i < N; i += blockDim.x) s[i] = sr[i];
  __syncthreads();
  for (int i = threadIdx.x; i < N; i += blockDim.x) {
    float si = s[i];
    int rank = 0;
    for (int j = 0; j < N; ++j) {
      float sj = s[j];
      rank += (sj > si) || (sj == si && j < i);
    }
    kp[i] = (rank < keep) ? 1 : 0;
  }
  __syncthreads();
  const float* fr = full + (size_t)row * Npad;
  for (int i = threadIdx.x; i < N; i += blockDim.x) {
    if (kp[i]) {
      int p = 0;
      for (int j = 0; j < i; ++j) p += kp[j];
      float v = fr[i];
      if (do_relu) v = fmaxf(v, 0.f);
      if (out_h) out_h[(size_t)row * keep + p] = (_Float16)v;
      if (out_f) out_f[(size_t)row * keep + p] = v;
    }
  }
}

// MoE-conv select: full is [B*64 spatial][256 ch]; keep 64 channels/batch,
// write ReLU'd f16 [B][64*64] in (kept-channel, spatial) order == reshape(B,-1).
__global__ void select_conv_kernel(const float* __restrict__ full,
                                   const float* __restrict__ scores,
                                   _Float16* __restrict__ out) {
  int b = blockIdx.x;
  __shared__ float s[256];
  __shared__ int   kp[256];
  __shared__ int   idxList[64];
  for (int i = threadIdx.x; i < 256; i += blockDim.x) s[i] = scores[(size_t)b * 256 + i];
  __syncthreads();
  for (int i = threadIdx.x; i < 256; i += blockDim.x) {
    float si = s[i];
    int rank = 0;
    for (int j = 0; j < 256; ++j) {
      float sj = s[j];
      rank += (sj > si) || (sj == si && j < i);
    }
    kp[i] = (rank < 64) ? 1 : 0;
  }
  __syncthreads();
  for (int i = threadIdx.x; i < 256; i += blockDim.x) {
    if (kp[i]) {
      int p = 0;
      for (int j = 0; j < i; ++j) p += kp[j];
      idxList[p] = i;
    }
  }
  __syncthreads();
  for (int t = threadIdx.x; t < 64 * 64; t += blockDim.x) {
    int rr = t >> 6, sp = t & 63;
    int ch = idxList[rr];
    float v = full[((size_t)(b * 64 + sp)) * 256 + ch];
    out[(size_t)b * 4096 + rr * 64 + sp] = (_Float16)fmaxf(v, 0.f);
  }
}

// ---------------------------------------------------------------------------
// Host-side orchestration
// ---------------------------------------------------------------------------
static inline void launch_gemm(const _Float16* A, const _Float16* W,
                               const float* bias, float* C,
                               int M, int Npad, int Nbias, int K, int mode,
                               hipStream_t stream) {
  int blocks = (M / 64) * (Npad / 16);  // 4 waves/block, 64x16 macro-tile
  wmma_gemm_kernel<<<blocks, 128, 0, stream>>>(A, W, bias, C, M, Npad, Nbias, K, mode);
}

extern "C" void kernel_launch(void* const* d_in, const int* in_sizes, int n_in,
                              void* d_out, int out_size, void* d_ws, size_t ws_size,
                              hipStream_t stream) {
  const int B = 1024;
  const float* x    = (const float*)d_in[0];
  const float* W0   = (const float*)d_in[1];
  const float* b0   = (const float*)d_in[2];
  const float* W2   = (const float*)d_in[3];
  const float* b2   = (const float*)d_in[4];
  const float* W4   = (const float*)d_in[5];
  const float* b4   = (const float*)d_in[6];
  const float* W6e  = (const float*)d_in[7];
  const float* b6e  = (const float*)d_in[8];
  const float* Wr6  = (const float*)d_in[9];
  const float* br6  = (const float*)d_in[10];
  const float* W9e  = (const float*)d_in[11];
  const float* b9e  = (const float*)d_in[12];
  const float* Wr9  = (const float*)d_in[13];
  const float* br9  = (const float*)d_in[14];
  const float* W11e = (const float*)d_in[15];
  const float* b11e = (const float*)d_in[16];
  const float* Wr11 = (const float*)d_in[17];
  const float* br11 = (const float*)d_in[18];
  const float* W13e = (const float*)d_in[19];
  const float* b13e = (const float*)d_in[20];
  const float* Wr13 = (const float*)d_in[21];
  const float* br13 = (const float*)d_in[22];

  char* ws = (char*)d_ws;
  size_t off = 0;
  auto alloc = [&](size_t bytes) -> char* {
    off = (off + 255) & ~(size_t)255;
    char* p = ws + off;
    off += bytes;
    return p;
  };

  float*    a1    = (float*)alloc(134217728ull);  // conv0 out; later aliased by im2col6
  float*    a2    = (float*)alloc(33554432ull);   // conv2 out
  float*    a3    = (float*)alloc(67108864ull);   // conv4 out
  _Float16* a3h   = (_Float16*)alloc(33554432ull);// conv4 out f16 (router6 input)
  _Float16* w6h   = (_Float16*)alloc(524288ull);  // [256][1024]
  _Float16* wr6h  = (_Float16*)alloc(8388608ull); // [256][16384]
  float*    c6    = (float*)alloc(67108864ull);   // conv6 full [65536][256]
  float*    s6    = (float*)alloc(1048576ull);    // scores6 [1024][256]
  _Float16* x9h   = (_Float16*)alloc(8388608ull); // [1024][4096]
  _Float16* w9h   = (_Float16*)alloc(16777216ull);// [2048][4096]
  _Float16* wr9h  = (_Float16*)alloc(16777216ull);
  float*    c9    = (float*)alloc(8388608ull);
  float*    s9    = (float*)alloc(8388608ull);
  _Float16* x11h  = (_Float16*)alloc(1048576ull); // [1024][512]
  _Float16* w11h  = (_Float16*)alloc(2097152ull); // [2048][512]
  _Float16* wr11h = (_Float16*)alloc(2097152ull);
  float*    c11   = (float*)alloc(8388608ull);
  float*    s11   = (float*)alloc(8388608ull);
  _Float16* x13h  = (_Float16*)alloc(1048576ull); // [1024][512]
  _Float16* w13h  = (_Float16*)alloc(49152ull);   // [48][512] (rows 40..47 zero)
  _Float16* wr13h = (_Float16*)alloc(49152ull);
  float*    c13   = (float*)alloc(196608ull);     // [1024][48]
  float*    s13   = (float*)alloc(196608ull);
  _Float16* col6  = (_Float16*)a1;                // im2col aliases dead a1 (134MB)

  // ---- front convs (fp32 direct) ----
  int t0 = B * 32 * 32 * 32;
  conv0_kernel<<<t0 / 256, 256, 0, stream>>>(x, W0, b0, a1, t0);
  int t2 = B * 32 * 16 * 16;
  conv2_kernel<<<t2 / 256, 256, 0, stream>>>(a1, W2, b2, a2, t2);
  int t4 = B * 64 * 16 * 16;
  conv4_kernel<<<t4 / 256, 256, 0, stream>>>(a2, W4, b4, a3, t4);

  // ---- stage 6: MoE conv (im2col + WMMA) + router + select ----
  int n3 = B * 16384;
  f32_to_f16_kernel<<<(n3 + 255) / 256, 256, 0, stream>>>(a3, a3h, n3);
  f32_to_f16_kernel<<<(262144 + 255) / 256, 256, 0, stream>>>(W6e, w6h, 262144);
  transpose_pad_f16_kernel<<<(256 * 16384) / 256, 256, 0, stream>>>(Wr6, wr6h, 16384, 256, 256);
  int tcol = 65536 * 1024;
  im2col6_kernel<<<tcol / 256, 256, 0, stream>>>(a3, col6, tcol);
  launch_gemm(col6, w6h, b6e, c6, 65536, 256, 256, 1024, GEMM_PLAIN, stream);
  launch_gemm(a3h, wr6h, br6, s6, 1024, 256, 256, 16384, GEMM_ABS, stream);
  select_conv_kernel<<<B, 256, 0, stream>>>(c6, s6, x9h);

  // ---- stage 9: MoE linear 4096 -> 4x512, keep 512 ----
  f32_to_f16_kernel<<<(8388608 + 255) / 256, 256, 0, stream>>>(W9e, w9h, 8388608);
  transpose_pad_f16_kernel<<<(2048 * 4096) / 256, 256, 0, stream>>>(Wr9, wr9h, 4096, 2048, 2048);
  launch_gemm(x9h, w9h, b9e, c9, B, 2048, 2048, 4096, GEMM_PLAIN, stream);
  launch_gemm(x9h, wr9h, br9, s9, B, 2048, 2048, 4096, GEMM_ABS, stream);
  select_linear_kernel<<<B, 256, 0, stream>>>(c9, s9, 2048, 2048, 512, 1, x11h, nullptr);

  // ---- stage 11: MoE linear 512 -> 4x512, keep 512 ----
  f32_to_f16_kernel<<<(1048576 + 255) / 256, 256, 0, stream>>>(W11e, w11h, 1048576);
  transpose_pad_f16_kernel<<<(2048 * 512) / 256, 256, 0, stream>>>(Wr11, wr11h, 512, 2048, 2048);
  launch_gemm(x11h, w11h, b11e, c11, B, 2048, 2048, 512, GEMM_PLAIN, stream);
  launch_gemm(x11h, wr11h, br11, s11, B, 2048, 2048, 512, GEMM_ABS, stream);
  select_linear_kernel<<<B, 256, 0, stream>>>(c11, s11, 2048, 2048, 512, 1, x13h, nullptr);

  // ---- stage 13: MoE linear 512 -> 4x10 (N padded 40->48), keep 10, no relu ----
  pad_rows_f16_kernel<<<(48 * 512) / 256, 256, 0, stream>>>(W13e, w13h, 512, 40, 48);
  transpose_pad_f16_kernel<<<(48 * 512) / 256, 256, 0, stream>>>(Wr13, wr13h, 512, 40, 48);
  launch_gemm(x13h, w13h, b13e, c13, B, 48, 40, 512, GEMM_PLAIN, stream);
  launch_gemm(x13h, wr13h, br13, s13, B, 48, 40, 512, GEMM_ABS, stream);
  select_linear_kernel<<<B, 256, 0, stream>>>(c13, s13, 40, 48, 10, 0, nullptr, (float*)d_out);

  (void)in_sizes; (void)n_in; (void)out_size; (void)ws_size;
}
